// NeuronSolarOpenMoE_62843961475294
// MI455X (gfx1250) — matrix operand order repaired and checked by reading, not compiled, not run on hardware
//
#include <hip/hip_runtime.h>
#include <hip/hip_bf16.h>

// ---------------- problem constants ----------------
#define NE 64      // experts
#define NK 8       // top-k
#define NH 1024    // hidden
#define NF 512     // expert ffn
#define NC 1024    // capacity
#define NT 4096    // tokens (B*S = 2*2048)

typedef __bf16 bf16_t;
typedef __attribute__((ext_vector_type(16))) __bf16 v16bf;
typedef __attribute__((ext_vector_type(8)))  __bf16 v8bf;
typedef __attribute__((ext_vector_type(4)))  __bf16 v4bf;
typedef __attribute__((ext_vector_type(8)))  float  v8f;

// ---------------- CDNA5 async global->LDS helpers ----------------
// GLOBAL_LOAD_ASYNC_TO_LDS_B128 (ASYNCcnt) with portable fallback.
// Builtin signature (from hipcc diagnostic): (int __vector(4) AS1*, int __vector(4) AS3*, imm, imm)
typedef int v4i_t __attribute__((vector_size(16)));
typedef __attribute__((address_space(1))) v4i_t* gptr_v4i;
typedef __attribute__((address_space(3))) v4i_t* lptr_v4i;

__device__ __forceinline__ void async_copy_b128(const void* g, void* l) {
#if defined(__has_builtin) && __has_builtin(__builtin_amdgcn_global_load_async_to_lds_b128)
    __builtin_amdgcn_global_load_async_to_lds_b128(
        (gptr_v4i)(unsigned long long)g,
        (lptr_v4i)(unsigned int)(unsigned long long)l,
        0, 0);
#else
    *(v8bf*)l = *(const v8bf*)g;
#endif
}
__device__ __forceinline__ void wait_async_all() {
#if defined(__has_builtin) && __has_builtin(__builtin_amdgcn_s_wait_asynccnt)
    __builtin_amdgcn_s_wait_asynccnt(0);
#else
    asm volatile("s_wait_asynccnt 0x0" ::: "memory");
#endif
}

// ---------------- WMMA fragment helpers (CDNA5 ISA §7.12.2) ----------------
// A (16x32 bf16) served from an LDS tile As[16][lda]: lane L holds row (L&15);
// kb = (L>>4)*16 -> elems 0..15 = K[kb..kb+15].
__device__ __forceinline__ v16bf load_a_frag_lds(const bf16_t* As, int lda, int k0, int lane) {
    const bf16_t* p = As + (lane & 15) * lda + k0 + ((lane >> 4) << 4);
    v8bf lo = *(const v8bf*)(p);       // ds_load_b128
    v8bf hi = *(const v8bf*)(p + 8);   // ds_load_b128
    return __builtin_shufflevector(lo, hi, 0,1,2,3,4,5,6,7,8,9,10,11,12,13,14,15);
}
// B (32x16 bf16) sourced from pre-transposed weight Bt[N][K] (K contiguous,
// stride ldb): lane L holds col (L&15), 16 contiguous K at k0 + (L>>4)*16.
__device__ __forceinline__ v16bf load_b_frag(const bf16_t* Bt, int ldb, int n0, int k0, int lane) {
    return *(const v16bf*)(Bt + (size_t)(n0 + (lane & 15)) * ldb + k0 + ((lane >> 4) << 4));
}

// ---------------- kernels ----------------
__global__ __launch_bounds__(64) void k_zero_counts(int* counts) {
    counts[threadIdx.x] = 0;
}

// Router logits in fp32 (router_logits is a checked output). Block per token.
__global__ __launch_bounds__(64) void k_router(const float* __restrict__ x,
                                               const float* __restrict__ wr,
                                               float* __restrict__ logits) {
    __shared__ __align__(16) float xs[NH];
    const int t = blockIdx.x;
    for (int i = threadIdx.x * 4; i < NH; i += 64 * 4)
        *(float4*)&xs[i] = *(const float4*)(x + (size_t)t * NH + i);
    __syncthreads();
    const float* w = wr + (size_t)threadIdx.x * NH;
    float acc = 0.f;
    for (int i = 0; i < NH; i += 4) {
        float4 wv = *(const float4*)(w + i);
        acc += wv.x * xs[i] + wv.y * xs[i + 1] + wv.z * xs[i + 2] + wv.w * xs[i + 3];
    }
    logits[(size_t)t * NE + threadIdx.x] = acc;
}

// Per-token sigmoid + bias selection, top-8 (ties -> lowest index), unbiased
// normalized combine weights. One thread per token.
__global__ __launch_bounds__(128) void k_topk(const float* __restrict__ logits,
                                              const float* __restrict__ bias,
                                              int* __restrict__ eidx,
                                              float* __restrict__ wts) {
    const int t = blockIdx.x * blockDim.x + threadIdx.x;
    if (t >= NT) return;
    float aff[NE], sel[NE];
    for (int e = 0; e < NE; ++e) {
        float a = 1.f / (1.f + __expf(-logits[(size_t)t * NE + e]));
        aff[e] = a;
        sel[e] = a + bias[e];
    }
    int   ci[NK];
    float cw[NK];
    float sum = 0.f;
    for (int k = 0; k < NK; ++k) {
        float best = -3.4e38f; int bi = 0;
        for (int e = 0; e < NE; ++e)
            if (sel[e] > best) { best = sel[e]; bi = e; }
        sel[bi] = -3.4e38f;
        ci[k] = bi; cw[k] = aff[bi]; sum += aff[bi];
    }
    float inv = 1.f / sum;
    for (int k = 0; k < NK; ++k) {
        eidx[t * NK + k] = ci[k];
        wts[t * NK + k]  = cw[k] * inv;
    }
}

// Slot assignment via atomics (capacity-checked).
__global__ __launch_bounds__(256) void k_assign(const int* __restrict__ eidx,
                                                int* __restrict__ counts,
                                                int* __restrict__ posr) {
    const int i = blockIdx.x * blockDim.x + threadIdx.x;
    if (i >= NT * NK) return;
    int p = atomicAdd(&counts[eidx[i]], 1);
    posr[i] = (p < NC) ? p : -1;
}

// Wave per (t,k): convert fp32 token row -> bf16 into buf[e][pos][:].
__global__ __launch_bounds__(256) void k_scatter(const float* __restrict__ x,
                                                 const int* __restrict__ eidx,
                                                 const int* __restrict__ posr,
                                                 bf16_t* __restrict__ buf) {
    const int pair = blockIdx.x * 8 + (threadIdx.x >> 5);
    const int lane = threadIdx.x & 31;
    const int p = posr[pair];
    if (p < 0) return;
    const int e = eidx[pair];
    const int t = pair / NK;
    const float* src = x + (size_t)t * NH;
    bf16_t* dst = buf + ((size_t)e * NC + p) * NH;
    for (int i = lane * 4; i < NH; i += 32 * 4) {
        float4 v = *(const float4*)(src + i);
        v4bf b;
        b[0] = (bf16_t)v.x; b[1] = (bf16_t)v.y; b[2] = (bf16_t)v.z; b[3] = (bf16_t)v.w;
        *(v4bf*)(dst + i) = b;
    }
}

// Generic per-expert 32x32 tiled transpose fp32 -> bf16: out[c][r] = in[r][c].
__global__ __launch_bounds__(256) void k_transpose_cvt(const float* __restrict__ in,
                                                       bf16_t* __restrict__ out,
                                                       int R, int Cc) {
    __shared__ float tile[32][33];
    const size_t eoff = (size_t)blockIdx.z * R * Cc;
    const int r0 = blockIdx.y * 32, c0 = blockIdx.x * 32;
    const int tx = threadIdx.x & 31, ty = threadIdx.x >> 5;
    for (int rr = ty; rr < 32; rr += 8)
        tile[rr][tx] = in[eoff + (size_t)(r0 + rr) * Cc + c0 + tx];
    __syncthreads();
    for (int cc = ty; cc < 32; cc += 8)
        out[eoff + (size_t)(c0 + cc) * R + r0 + tx] = (bf16_t)tile[tx][cc];
}

// Gate+Up GEMM + fused SiLU*Up -> act (bf16).
// Block: 16 rows x 256 cols; A tile (16xNH, contiguous 32KiB) async-staged to
// LDS once, shared by 8 waves. Each wave: two 16x16 N-tiles -> 4 WMMAs per A frag.
__global__ __launch_bounds__(256) void k_gemm_gateup(const bf16_t* __restrict__ buf,
                                                     const bf16_t* __restrict__ wgt,
                                                     const bf16_t* __restrict__ wut,
                                                     bf16_t* __restrict__ act,
                                                     const int* __restrict__ counts) {
    __shared__ __align__(16) bf16_t As[16 * NH];   // 32 KiB
    const int e  = blockIdx.z;
    const int m0 = blockIdx.y * 16;
    int cnt = counts[e]; if (cnt > NC) cnt = NC;
    if (m0 >= cnt) return;                          // block-uniform exit
    const bf16_t* A = buf + ((size_t)e * NC + m0) * NH;   // contiguous 16xNH tile
    for (int c = threadIdx.x * 8; c < 16 * NH; c += 256 * 8)
        async_copy_b128(A + c, &As[c]);
    wait_async_all();
    __syncthreads();

    const int lane = threadIdx.x & 31;
    const int n0   = blockIdx.x * 256 + (threadIdx.x >> 5) * 32;
    const bf16_t* Bg = wgt + (size_t)e * NF * NH;  // [F][H]
    const bf16_t* Bu = wut + (size_t)e * NF * NH;  // [F][H]
    v8f cg0 = {}, cu0 = {}, cg1 = {}, cu1 = {};
    for (int k0 = 0; k0 < NH; k0 += 32) {
        if (k0 + 32 < NH) {
            __builtin_prefetch(Bg + (size_t)(n0 + (lane & 15)) * NH + k0 + 32, 0, 3);
            __builtin_prefetch(Bu + (size_t)(n0 + (lane & 15)) * NH + k0 + 32, 0, 3);
        }
        v16bf a   = load_a_frag_lds(As, NH, k0, lane);
        v16bf bg0 = load_b_frag(Bg, NH, n0,      k0, lane);
        v16bf bu0 = load_b_frag(Bu, NH, n0,      k0, lane);
        v16bf bg1 = load_b_frag(Bg, NH, n0 + 16, k0, lane);
        v16bf bu1 = load_b_frag(Bu, NH, n0 + 16, k0, lane);
        cg0 = __builtin_amdgcn_wmma_f32_16x16x32_bf16(false, a, false, bg0, (short)0, cg0, false, false);
        cu0 = __builtin_amdgcn_wmma_f32_16x16x32_bf16(false, a, false, bu0, (short)0, cu0, false, false);
        cg1 = __builtin_amdgcn_wmma_f32_16x16x32_bf16(false, a, false, bg1, (short)0, cg1, false, false);
        cu1 = __builtin_amdgcn_wmma_f32_16x16x32_bf16(false, a, false, bu1, (short)0, cu1, false, false);
    }
    // D layout: row = m0 + v + 8*(lane>>4), col = n0 + (lane&15)
    const int col = n0 + (lane & 15);
    const int mb  = m0 + ((lane >> 4) << 3);
    bf16_t* out = act + (size_t)e * NC * NF;
#pragma unroll
    for (int v = 0; v < 8; ++v) {
        float g0 = cg0[v], u0 = cu0[v];
        float g1 = cg1[v], u1 = cu1[v];
        out[(size_t)(mb + v) * NF + col]      = (bf16_t)((g0 / (1.f + __expf(-g0))) * u0);
        out[(size_t)(mb + v) * NF + col + 16] = (bf16_t)((g1 / (1.f + __expf(-g1))) * u1);
    }
}

// Down projection: o = act @ w_down (bf16 out). Same LDS-staged structure.
__global__ __launch_bounds__(256) void k_gemm_down(const bf16_t* __restrict__ act,
                                                   const bf16_t* __restrict__ wdt,
                                                   bf16_t* __restrict__ o,
                                                   const int* __restrict__ counts) {
    __shared__ __align__(16) bf16_t As[16 * NF];   // 16 KiB
    const int e  = blockIdx.z;
    const int m0 = blockIdx.y * 16;
    int cnt = counts[e]; if (cnt > NC) cnt = NC;
    if (m0 >= cnt) return;
    const bf16_t* A = act + ((size_t)e * NC + m0) * NF;   // contiguous 16xNF tile
    for (int c = threadIdx.x * 8; c < 16 * NF; c += 256 * 8)
        async_copy_b128(A + c, &As[c]);
    wait_async_all();
    __syncthreads();

    const int lane = threadIdx.x & 31;
    const int n0   = blockIdx.x * 256 + (threadIdx.x >> 5) * 32;
    const bf16_t* Bt = wdt + (size_t)e * NH * NF;  // [H][F]
    v8f c0 = {}, c1 = {};
    for (int k0 = 0; k0 < NF; k0 += 32) {
        if (k0 + 32 < NF)
            __builtin_prefetch(Bt + (size_t)(n0 + (lane & 15)) * NF + k0 + 32, 0, 3);
        v16bf a  = load_a_frag_lds(As, NF, k0, lane);
        v16bf b0 = load_b_frag(Bt, NF, n0,      k0, lane);
        v16bf b1 = load_b_frag(Bt, NF, n0 + 16, k0, lane);
        c0 = __builtin_amdgcn_wmma_f32_16x16x32_bf16(false, a, false, b0, (short)0, c0, false, false);
        c1 = __builtin_amdgcn_wmma_f32_16x16x32_bf16(false, a, false, b1, (short)0, c1, false, false);
    }
    const int col = n0 + (lane & 15);
    const int mb  = m0 + ((lane >> 4) << 3);
    bf16_t* out = o + (size_t)e * NC * NH;
#pragma unroll
    for (int v = 0; v < 8; ++v) {
        out[(size_t)(mb + v) * NH + col]      = (bf16_t)c0[v];
        out[(size_t)(mb + v) * NH + col + 16] = (bf16_t)c1[v];
    }
}

// y[t] = sum_k w[t,k] * o[e_k, pos_k]. Block per token, thread covers 4 h.
__global__ __launch_bounds__(256) void k_combine(const bf16_t* __restrict__ o,
                                                 const int* __restrict__ eidx,
                                                 const int* __restrict__ posr,
                                                 const float* __restrict__ wts,
                                                 float* __restrict__ y) {
    const int t  = blockIdx.x;
    const int h0 = threadIdx.x * 4;
    float4 acc = make_float4(0.f, 0.f, 0.f, 0.f);
#pragma unroll
    for (int k = 0; k < NK; ++k) {
        const int p = posr[t * NK + k];
        if (p < 0) continue;
        const int e = eidx[t * NK + k];
        const float w = wts[t * NK + k];
        v4bf v = *(const v4bf*)(o + ((size_t)e * NC + p) * NH + h0);
        acc.x += w * (float)v[0];
        acc.y += w * (float)v[1];
        acc.z += w * (float)v[2];
        acc.w += w * (float)v[3];
    }
    *(float4*)(y + (size_t)t * NH + h0) = acc;
}

// ---------------- host launcher ----------------
extern "C" void kernel_launch(void* const* d_in, const int* in_sizes, int n_in,
                              void* d_out, int out_size, void* d_ws, size_t ws_size,
                              hipStream_t stream) {
    const float* x     = (const float*)d_in[0];   // [B,S,H]
    const float* wr    = (const float*)d_in[1];   // [E,H]
    const float* bias  = (const float*)d_in[2];   // [E]
    const float* wgate = (const float*)d_in[3];   // [E,H,F]
    const float* wup   = (const float*)d_in[4];   // [E,H,F]
    const float* wdown = (const float*)d_in[5];   // [E,F,H]

    // outputs concatenated flat: y [T,H] f32 | router_logits [T,E] f32 | expert_index [T,K] i32
    float* y      = (float*)d_out;
    float* logits = y + (size_t)NT * NH;
    int*   eidx   = (int*)(logits + (size_t)NT * NE);

    // workspace layout (256B aligned slabs)
    char* ws = (char*)d_ws;
    size_t off = 0;
    auto alloc = [&](size_t bytes) { char* p = ws + off; off = (off + bytes + 255) & ~(size_t)255; return p; };
    int*    counts = (int*)   alloc(NE * sizeof(int));
    int*    posr   = (int*)   alloc((size_t)NT * NK * sizeof(int));
    float*  wts    = (float*) alloc((size_t)NT * NK * sizeof(float));
    bf16_t* buf    = (bf16_t*)alloc((size_t)NE * NC * NH * sizeof(bf16_t));   // 128 MiB
    bf16_t* act    = (bf16_t*)alloc((size_t)NE * NC * NF * sizeof(bf16_t));   //  64 MiB
    bf16_t* obuf   = (bf16_t*)alloc((size_t)NE * NC * NH * sizeof(bf16_t));   // 128 MiB
    bf16_t* wgt    = (bf16_t*)alloc((size_t)NE * NF * NH * sizeof(bf16_t));   //  64 MiB
    bf16_t* wut    = (bf16_t*)alloc((size_t)NE * NF * NH * sizeof(bf16_t));   //  64 MiB
    bf16_t* wdt    = (bf16_t*)alloc((size_t)NE * NH * NF * sizeof(bf16_t));   //  64 MiB
    (void)ws_size; (void)in_sizes; (void)n_in; (void)out_size;

    // 1) reset per-expert counters (atomics must start at 0 every replay)
    k_zero_counts<<<1, NE, 0, stream>>>(counts);
    // 2) router logits (fp32, checked output)
    k_router<<<NT, 64, 0, stream>>>(x, wr, logits);
    // 3) top-8 selection + combine weights
    k_topk<<<NT / 128, 128, 0, stream>>>(logits, bias, eidx, wts);
    // 4) capacity-bucketed dispatch
    k_assign<<<(NT * NK) / 256, 256, 0, stream>>>(eidx, counts, posr);
    k_scatter<<<(NT * NK) / 8, 256, 0, stream>>>(x, eidx, posr, buf);
    // 5) weight prep: transpose+convert so B-fragments are contiguous bf16
    k_transpose_cvt<<<dim3(NF / 32, NH / 32, NE), 256, 0, stream>>>(wgate, wgt, NH, NF); // [H,F]->[F,H]
    k_transpose_cvt<<<dim3(NF / 32, NH / 32, NE), 256, 0, stream>>>(wup,   wut, NH, NF); // [H,F]->[F,H]
    k_transpose_cvt<<<dim3(NH / 32, NF / 32, NE), 256, 0, stream>>>(wdown, wdt, NF, NH); // [F,H]->[H,F]
    // 6) expert GEMMs on the matrix units (A tiles async-staged to LDS)
    k_gemm_gateup<<<dim3(NF / 256, NC / 16, NE), 256, 0, stream>>>(buf, wgt, wut, act, counts);
    k_gemm_down  <<<dim3(NH / 256, NC / 16, NE), 256, 0, stream>>>(act, wdt, obuf, counts);
    // 7) weighted combine
    k_combine<<<NT, 256, 0, stream>>>(obuf, eidx, posr, wts, y);
}
